// NNLoss_50173807951931
// MI455X (gfx1250) — matrix operand order repaired and checked by reading, not compiled, not run on hardware
//
#include <hip/hip_runtime.h>
#include <hip/hip_bf16.h>
#include <math.h>

typedef __bf16 bf16;
typedef __attribute__((ext_vector_type(16))) __bf16 v16bf;
typedef __attribute__((ext_vector_type(8)))  __bf16 v8bf;
typedef __attribute__((ext_vector_type(8)))  float  v8f;
typedef __attribute__((ext_vector_type(4)))  unsigned int u32x4;
typedef __attribute__((ext_vector_type(8)))  int i32x8;
typedef __attribute__((ext_vector_type(4)))  int i32x4;

__device__ __forceinline__ bf16 f2bf(float f) {
  union { float f; unsigned u; } v; v.f = f;
  unsigned r = v.u + 0x7fffu + ((v.u >> 16) & 1u);   // round-to-nearest-even
  union { unsigned short s; bf16 b; } o; o.s = (unsigned short)(r >> 16);
  return o.b;
}
__device__ __forceinline__ float bf2f(bf16 b) {
  union { unsigned short s; bf16 b; } i; i.b = b;
  union { unsigned u; float f; } o; o.u = ((unsigned)i.s) << 16;
  return o.f;
}

// ---------------------------------------------------------------------------
// conv1_1 (C_in=3): direct fp32 conv fused with ImageNet mean/std normalize.
// in:  NCHW fp32 [B,3,H,W]  -> out: NHWC bf16 [B,H,W,64], ReLU applied.
// ---------------------------------------------------------------------------
__global__ void conv1_direct_k(const float* __restrict__ in,
                               const float* __restrict__ w,     // [64][3][3][3] OIHW
                               const float* __restrict__ bias,
                               bf16* __restrict__ out,
                               int NB, int H, int W)
{
  const float mean[3] = {0.485f, 0.456f, 0.406f};
  const float istd[3] = {1.0f/0.229f, 1.0f/0.224f, 1.0f/0.225f};
  long long idx = (long long)blockIdx.x * blockDim.x + threadIdx.x;
  long long total = (long long)NB * H * W * 64;
  if (idx >= total) return;
  int o = (int)(idx & 63);
  long long p = idx >> 6;                 // b*H*W + y*W + x
  int x = (int)(p % W);
  int y = (int)((p / W) % H);
  int b = (int)(p / ((long long)W * H));
  float acc = bias[o];
  #pragma unroll
  for (int ch = 0; ch < 3; ++ch) {
    const float* ic = in + ((size_t)(b * 3 + ch) * H) * W;
    const float* wc = w + (size_t)(o * 3 + ch) * 9;
    #pragma unroll
    for (int kh = 0; kh < 3; ++kh) {
      int iy = y + kh - 1;
      if ((unsigned)iy >= (unsigned)H) continue;
      #pragma unroll
      for (int kw = 0; kw < 3; ++kw) {
        int ix = x + kw - 1;
        if ((unsigned)ix >= (unsigned)W) continue;
        float v = (ic[(size_t)iy * W + ix] - mean[ch]) * istd[ch];
        acc += v * wc[kh * 3 + kw];
      }
    }
  }
  out[(size_t)p * 64 + o] = f2bf(acc > 0.0f ? acc : 0.0f);
}

// ---------------------------------------------------------------------------
// K-loop of the implicit-GEMM conv. GUARD=false is the interior fast path:
// no exec masking, no zero-fill, pure global_load_b128 + v_wmma.
// ---------------------------------------------------------------------------
template <bool GUARD>
__device__ __forceinline__ void conv_kloop(v8f& acc,
    const bf16* __restrict__ ibat, const bf16* __restrict__ wbase,
    int CIN, int inStride, int y, int x, int H, int W, int g)
{
  for (int kh = 0; kh < 3; ++kh) {
    const int iy = y + kh - 1;
    const bool yok = !GUARD || ((unsigned)iy < (unsigned)H);
    for (int kw = 0; kw < 3; ++kw) {
      const int ix = x + kw - 1;
      const bool ok = !GUARD || (yok && ((unsigned)ix < (unsigned)W));
      const bf16* ip = ibat + ((size_t)iy * W + ix) * inStride;
      const bf16* wp = wbase + (kh * 3 + kw) * CIN;
      for (int c0 = 0; c0 < CIN; c0 += 32) {
        // A (16x32 bf16): lane m holds K = 16g..16g+15
        v8bf alo = *(const v8bf*)(wp + c0 + 16 * g);
        v8bf ahi = *(const v8bf*)(wp + c0 + 16 * g + 8);
        v16bf afrag = __builtin_shufflevector(alo, ahi,
            0,1,2,3,4,5,6,7,8,9,10,11,12,13,14,15);
        // B (32x16 bf16): lane col n holds K = 16g..16g+15 (contiguous NHWC)
        v16bf bfrag;
        if (GUARD) {
          v16bf z = {};
          bfrag = z;
          if (ok) bfrag = *(const v16bf*)(ip + c0 + 16 * g);
        } else {
          bfrag = *(const v16bf*)(ip + c0 + 16 * g);
        }
        acc = __builtin_amdgcn_wmma_f32_16x16x32_bf16(
            false, afrag, false, bfrag, (short)0, acc, false, false);
      }
    }
  }
}

// ---------------------------------------------------------------------------
// Generic 3x3 SAME conv as implicit GEMM on v_wmma_f32_16x16x32_bf16.
// One wave computes a 16-outC x 16-pixel tile; K loop over (kh,kw,c32).
// Activations NHWC bf16; weights pre-transformed to [O][kh][kw][I] bf16.
// Bias + ReLU fused into the accumulator store.
// ---------------------------------------------------------------------------
__global__ void __launch_bounds__(128)
conv3x3_wmma_k(const bf16* __restrict__ in, int inStride, int CIN,
               const bf16* __restrict__ wt, const float* __restrict__ bias,
               bf16* __restrict__ out, int outStride, int COUT,
               int NB, int H, int W)
{
  const int lane = threadIdx.x & 31;
  const int wave = blockIdx.x * (blockDim.x >> 5) + (threadIdx.x >> 5);
  const int otiles = COUT >> 4;
  const int ptiles = (H * W) >> 4;
  const int ot = wave % otiles;
  const int tmp = wave / otiles;
  const int pt = tmp % ptiles;
  const int b  = tmp / ptiles;
  if (b >= NB) return;

  const int n = lane & 15;        // A row (out channel) AND B column (pixel)
  const int g = lane >> 4;        // lane half selects K sub-chunk
  const int p = pt * 16 + n;      // W % 16 == 0 -> tile never crosses a row
  const int y = p / W;
  const int x = p - y * W;
  const int x0 = x - n;           // wave-uniform tile origin

  v8f acc = {};
  const bf16* wbase = wt + (size_t)(ot * 16 + n) * 9 * CIN;
  const bf16* ibat  = in + (size_t)b * H * W * inStride;

  // Uniform branch: ~87% of tiles need no boundary masking at all.
  const bool interior = (y >= 1) && (y < H - 1) && (x0 > 0) && (x0 + 16 < W);
  if (interior)
    conv_kloop<false>(acc, ibat, wbase, CIN, inStride, y, x, H, W, g);
  else
    conv_kloop<true>(acc, ibat, wbase, CIN, inStride, y, x, H, W, g);

  // D layout: lane half g, VGPR r -> out channel 8g+r; column n -> pixel p.
  const int obase = ot * 16 + 8 * g;
  v8bf ov;
  #pragma unroll
  for (int r = 0; r < 8; ++r) {
    float v = acc[r] + bias[obase + r];
    ov[r] = f2bf(v > 0.0f ? v : 0.0f);
  }
  bf16* op = out + ((size_t)b * H * W + p) * outStride + obase;
  *(v8bf*)op = ov;
}

// ---------------------------------------------------------------------------
// 2x2 stride-2 max pool, NHWC bf16.
// ---------------------------------------------------------------------------
__global__ void maxpool2_k(const bf16* __restrict__ in, bf16* __restrict__ out,
                           int NB, int H, int W, int C)
{
  const int H2 = H >> 1, W2 = W >> 1;
  long long idx = (long long)blockIdx.x * blockDim.x + threadIdx.x;
  long long total = (long long)NB * H2 * W2 * C;
  if (idx >= total) return;
  int c = (int)(idx % C);
  long long t = idx / C;
  int x = (int)(t % W2);
  int y = (int)((t / W2) % H2);
  int b = (int)(t / ((long long)W2 * H2));
  const bf16* ib = in + (((size_t)b * H + 2 * y) * W + 2 * x) * C + c;
  float m = fmaxf(fmaxf(bf2f(ib[0]), bf2f(ib[C])),
                  fmaxf(bf2f(ib[(size_t)W * C]), bf2f(ib[(size_t)W * C + C])));
  out[idx] = f2bf(m);
}

// ---------------------------------------------------------------------------
// Per-row inverse norm: inv[row] = 1/(||f_row|| + 1e-8). One wave per row.
// ---------------------------------------------------------------------------
__global__ void rownorm_k(const bf16* __restrict__ f, float* __restrict__ inv,
                          int rows, int C)
{
  const int wave = blockIdx.x * (blockDim.x >> 5) + (threadIdx.x >> 5);
  const int lane = threadIdx.x & 31;
  if (wave >= rows) return;
  const bf16* rp = f + (size_t)wave * C;
  float s = 0.0f;
  for (int c = lane; c < C; c += 32) {
    float v = bf2f(rp[c]);
    s += v * v;
  }
  #pragma unroll
  for (int off = 16; off > 0; off >>= 1) s += __shfl_xor(s, off, 32);
  if (lane == 0) inv[wave] = 1.0f / (sqrtf(s) + 1e-8f);
}

// ---------------------------------------------------------------------------
// Fused cosine-similarity GEMM + running max:  loss = mean(1 - max_m xn.sn_m)
// (the gather+renormalize in the reference collapses to the max sim itself).
//
// Block = 4 waves = 4 x-tiles sharing one 16x768 style tile per M-step.
// The style tile is one contiguous 24 KB span of sfeat, DMA'd into LDS by the
// Tensor Data Mover (1-D D#: data_size=2B, tensor_dim0=tile_dim0=16*C elems),
// synchronized with s_wait_tensorcnt + workgroup barriers; A-fragments then
// come from LDS (ds_load_b128), cutting global A-traffic 4x.
// ---------------------------------------------------------------------------
__global__ void __launch_bounds__(128)
simmax_k(const bf16* __restrict__ xf, const bf16* __restrict__ sf,
         const float* __restrict__ invx, const float* __restrict__ invs,
         float* __restrict__ loss, int NB, int N, int C)
{
  __shared__ bf16 stile[16 * 768];

  const int lane = threadIdx.x & 31;
  const int wv   = threadIdx.x >> 5;          // 0..3: x-tile within block
  const int ntiles  = N >> 4;                 // 256
  const int ngroups = ntiles >> 2;            // 64 blocks per batch image
  const int b  = blockIdx.x / ngroups;
  const int nt = (blockIdx.x % ngroups) * 4 + wv;
  if (b >= NB) return;

  const int n = lane & 15;
  const int g = lane >> 4;
  const int xn = nt * 16 + n;
  const float ivx = invx[b * N + xn];
  const bf16* xp = xf + ((size_t)b * N + xn) * C;
  float runmax = -3.0e38f;

  // Low 32 bits of the flat shared address = LDS byte offset (aperture map).
  const unsigned ldsOff = (unsigned)(uintptr_t)(void*)&stile[0];

  for (int mt = 0; mt < ntiles; ++mt) {
    __syncthreads();                          // previous tile fully consumed
    if (wv == 0) {
      const bf16* gsrc = sf + ((size_t)b * N + mt * 16) * C;  // contiguous
      const unsigned long long ga = (unsigned long long)(uintptr_t)gsrc;
      const unsigned elems = 16u * (unsigned)C;               // 12288 x 2B
      u32x4 g0;
      g0[0] = 1u;                                             // count=1, user D#
      g0[1] = ldsOff;                                         // lds_addr (bytes)
      g0[2] = (unsigned)(ga & 0xffffffffu);                   // global_addr[31:0]
      g0[3] = (unsigned)((ga >> 32) & 0x01ffffffu) | (2u << 30); // [56:32] | type=2
      i32x8 g1;
      g1[0] = (int)(1u << 16);          // workgroup_mask=0, data_size=1 (2 B)
      g1[1] = (int)(elems << 16);       // tensor_dim0[15:0] << 16
      g1[2] = (int)(1u << 16);          // tensor_dim0 hi=0, tensor_dim1=1
      g1[3] = (int)(elems << 16);       // tile_dim0 = 16*C
      g1[4] = 1;                        // tile_dim1=1, tile_dim2=0
      g1[5] = (int)elems;               // tensor_dim0_stride
      g1[6] = 0;
      g1[7] = 0;
      i32x4 zz = {0, 0, 0, 0};          // groups 2/3 unused (<=2-D tensor)
      i32x8 z8 = {0, 0, 0, 0, 0, 0, 0, 0};
      __builtin_amdgcn_tensor_load_to_lds(g0, g1, zz, zz, z8, 0);
      __builtin_amdgcn_s_wait_tensorcnt(0);
    }
    __syncthreads();                          // tile visible to all 4 waves

    v8f acc = {};
    const bf16* sp = stile + n * C;           // A row m=n from LDS
    for (int c0 = 0; c0 < C; c0 += 32) {
      v8bf alo = *(const v8bf*)(sp + c0 + 16 * g);
      v8bf ahi = *(const v8bf*)(sp + c0 + 16 * g + 8);
      v16bf afrag = __builtin_shufflevector(alo, ahi,
          0,1,2,3,4,5,6,7,8,9,10,11,12,13,14,15);
      v16bf bfrag = *(const v16bf*)(xp + c0 + 16 * g);
      acc = __builtin_amdgcn_wmma_f32_16x16x32_bf16(
          false, afrag, false, bfrag, (short)0, acc, false, false);
    }
    const float* ivsp = invs + b * N + mt * 16 + 8 * g;
    #pragma unroll
    for (int r = 0; r < 8; ++r)
      runmax = fmaxf(runmax, acc[r] * ivsp[r] * ivx);
  }
  // lane L and L^16 hold disjoint style halves for the same x column
  runmax = fmaxf(runmax, __shfl_xor(runmax, 16, 32));
  float contrib = (lane < 16) ? (1.0f - runmax) : 0.0f;
  #pragma unroll
  for (int off = 16; off > 0; off >>= 1) contrib += __shfl_xor(contrib, off, 32);
  if (lane == 0) atomicAdd(loss, contrib / (float)(NB * N));
}

// ---------------------------------------------------------------------------
// Weight transform [O][I][3][3] f32 -> [O][kh][kw][I] bf16 (A-fragment layout)
// ---------------------------------------------------------------------------
__global__ void wxform_k(const float* __restrict__ w, bf16* __restrict__ wt,
                         int O, int I)
{
  int idx = blockIdx.x * blockDim.x + threadIdx.x;
  int total = O * 9 * I;
  if (idx >= total) return;
  int i = idx % I;
  int k = (idx / I) % 9;
  int o = idx / (9 * I);
  wt[idx] = f2bf(w[((size_t)o * I + i) * 9 + k]);
}

__global__ void zero_k(float* p, int n) {
  int i = blockIdx.x * blockDim.x + threadIdx.x;
  if (i < n) p[i] = 0.0f;
}

// ---------------------------------------------------------------------------
extern "C" void kernel_launch(void* const* d_in, const int* in_sizes, int n_in,
                              void* d_out, int out_size, void* d_ws, size_t ws_size,
                              hipStream_t stream)
{
  (void)in_sizes; (void)n_in; (void)ws_size;
  const float* outputs = (const float*)d_in[0];
  const float* styles  = (const float*)d_in[1];
  const float* w[7]; const float* bb[7];
  for (int i = 0; i < 7; ++i) { w[i] = (const float*)d_in[2 + 2 * i];
                                bb[i] = (const float*)d_in[3 + 2 * i]; }

  const int NB = 4;
  char* ws = (char*)d_ws;
  size_t off = 0;
  auto alloc = [&](size_t bytes) -> void* {
    void* p = ws + off;
    off = (off + bytes + 255) & ~(size_t)255;
    return p;
  };
  bf16* bufA  = (bf16*)alloc((size_t)NB * 256 * 256 * 64 * 2);  // 33.5 MB
  bf16* bufB  = (bf16*)alloc((size_t)NB * 256 * 256 * 64 * 2);  // 33.5 MB
  bf16* xfeat = (bf16*)alloc((size_t)NB * 4096 * 768 * 2);      // 25.2 MB
  bf16* sfeat = (bf16*)alloc((size_t)NB * 4096 * 768 * 2);      // 25.2 MB
  float* invx = (float*)alloc((size_t)NB * 4096 * 4);
  float* invs = (float*)alloc((size_t)NB * 4096 * 4);
  const int Oc[7] = {64, 64, 128, 128, 256, 256, 256};
  const int Ic[7] = {3, 64, 64, 128, 128, 256, 256};
  bf16* wt[7] = {nullptr};
  for (int i = 1; i < 7; ++i) wt[i] = (bf16*)alloc((size_t)Oc[i] * 9 * Ic[i] * 2);

  zero_k<<<1, 32, 0, stream>>>((float*)d_out, out_size);
  for (int i = 1; i < 7; ++i) {
    int tot = Oc[i] * 9 * Ic[i];
    wxform_k<<<(tot + 255) / 256, 256, 0, stream>>>(w[i], wt[i], Oc[i], Ic[i]);
  }

  auto convw = [&](const bf16* in, int inStride, int CIN, int wi,
                   bf16* out, int outStride, int COUT, int H, int W) {
    int tiles = NB * ((H * W) / 16) * (COUT / 16);     // divisible by 4 (NB=4)
    conv3x3_wmma_k<<<tiles / 4, 128, 0, stream>>>(in, inStride, CIN, wt[wi], bb[wi],
                                                  out, outStride, COUT, NB, H, W);
  };

  auto pipeline = [&](const float* img, bf16* feat, float* inv) {
    { long long tot = (long long)NB * 256 * 256 * 64;
      conv1_direct_k<<<(int)(tot / 256), 256, 0, stream>>>(img, w[0], bb[0],
                                                           bufA, NB, 256, 256); }
    convw(bufA, 64, 64, 1, bufB, 64, 64, 256, 256);            // conv1_2
    { long long tot = (long long)NB * 128 * 128 * 64;
      maxpool2_k<<<(int)((tot + 255) / 256), 256, 0, stream>>>(bufB, bufA,
                                                               NB, 256, 256, 64); }
    convw(bufA, 64, 64, 2, bufB, 128, 128, 128, 128);          // conv2_1
    convw(bufB, 128, 128, 3, bufA, 128, 128, 128, 128);        // conv2_2
    { long long tot = (long long)NB * 64 * 64 * 128;
      maxpool2_k<<<(int)((tot + 255) / 256), 256, 0, stream>>>(bufA, bufB,
                                                               NB, 128, 128, 128); }
    // conv3_x write straight into channel slices of the concat buffer.
    convw(bufB, 128, 128, 4, feat + 0,   768, 256, 64, 64);    // conv3_1 -> f1
    convw(feat + 0,   768, 256, 5, feat + 256, 768, 256, 64, 64); // conv3_2 -> f2
    convw(feat + 256, 768, 256, 6, feat + 512, 768, 256, 64, 64); // conv3_3 -> f3
    rownorm_k<<<(NB * 4096) / 4, 128, 0, stream>>>(feat, inv, NB * 4096, 768);
  };

  pipeline(outputs, xfeat, invx);
  pipeline(styles,  sfeat, invs);

  simmax_k<<<(NB * 4096 / 16) / 4, 128, 0, stream>>>(xfeat, sfeat, invx, invs,
                                                     (float*)d_out, NB, 4096, 768);
}
